// MemoryNextTextModel_42245298323761
// MI455X (gfx1250) — matrix operand order repaired and checked by reading, not compiled, not run on hardware
//
#include <hip/hip_runtime.h>

// ---------------------------------------------------------------------------
// CDNA5 (gfx1250) implementation of MemoryNextTextModel forward.
// All matmuls use v_wmma_f32_16x16x32_f16 (f16 inputs, f32 accumulate).
// ---------------------------------------------------------------------------

typedef __attribute__((ext_vector_type(16))) _Float16 v16h;
typedef __attribute__((ext_vector_type(8)))  float    v8f;

union Frag16 { v16h v; unsigned u[8]; };

__device__ __forceinline__ float sigmoidf_(float x) {
  return 1.0f / (1.0f + __expf(-x));
}

// --------------------------- small utility kernels -------------------------

__global__ __launch_bounds__(256) void k_cvt_f16(const float* __restrict__ s,
                                                 _Float16* __restrict__ d, int n) {
  int i = blockIdx.x * 256 + threadIdx.x;
  if (i < n) d[i] = (_Float16)s[i];
}

__global__ __launch_bounds__(256) void k_fill0_f32(float* __restrict__ p, int n) {
  int i = blockIdx.x * 256 + threadIdx.x;
  if (i < n) p[i] = 0.0f;
}

__global__ __launch_bounds__(256) void k_fill0_f16(_Float16* __restrict__ p, int n) {
  int i = blockIdx.x * 256 + threadIdx.x;
  if (i < n) p[i] = (_Float16)0.0f;
}

// dst[row, c] = (f16) tab[ids[row], c]
__global__ __launch_bounds__(256) void k_gather_f16(const int* __restrict__ ids,
                                                    const float* __restrict__ tab,
                                                    _Float16* __restrict__ dst,
                                                    int total, int dim) {
  int i = blockIdx.x * 256 + threadIdx.x;
  if (i >= total) return;
  int row = i / dim;
  int c   = i - row * dim;
  dst[i] = (_Float16)tab[(size_t)ids[row] * dim + c];
}

// decoder input gather: row = b*(T-1)+t ; id = tgt[b*T + t]
__global__ __launch_bounds__(256) void k_gather_tgt_f16(const int* __restrict__ tgt,
                                                        const float* __restrict__ tab,
                                                        _Float16* __restrict__ dst,
                                                        int Tm1, int Tfull, int dim,
                                                        int total) {
  int i = blockIdx.x * 256 + threadIdx.x;
  if (i >= total) return;
  int row = i / dim;
  int c   = i - row * dim;
  int b   = row / Tm1;
  int t   = row - b * Tm1;
  dst[i] = (_Float16)tab[(size_t)tgt[b * Tfull + t] * dim + c];
}

// fin[row, 0:dm) = h16 ; fin[row, dm:dm+imgd) = (f16) img
__global__ __launch_bounds__(256) void k_build_fin(const _Float16* __restrict__ h16,
                                                   const float* __restrict__ img,
                                                   _Float16* __restrict__ fin,
                                                   int rows, int dm, int imgd) {
  int w = dm + imgd;
  int total = rows * w;
  int i = blockIdx.x * 256 + threadIdx.x;
  if (i >= total) return;
  int row = i / w;
  int c   = i - row * w;
  fin[i] = (c < dm) ? h16[(size_t)row * dm + c]
                    : (_Float16)img[(size_t)row * imgd + (c - dm)];
}

// --------------------------- WMMA GEMM ------------------------------------
// C[M,N] (f32) = A[M,K] (f16) @ B[N,K]^T (f16) + bias[N]; optional tanh.
// Tile 64x64, 256 threads = 8 waves; each wave: 1 m-tile x 2 n-tiles.
// N, K must be multiples of 64/32; M is edge-guarded.
//
// Fragment packing (per CDNA5 ISA 16-bit layouts, wave32):
//   A (16x32, MxK): lane l -> row M = l&15; khalf = (l>=16)?8:0;
//                   VGPR i (i<4): K pair = khalf + 2*(i&3)
//                   VGPR i (i>=4): K pair = 16 + khalf + 2*(i&3)
//   B (32x16, KxN): lane l -> col N = l&15; khalf = (l>=16)?8:0;
//                   VGPR i: same K pairs as A (khalf + 2*(i&3), +16 for i>=4)

__global__ __launch_bounds__(256) void k_gemm_f16(const _Float16* __restrict__ A,
                                                  const _Float16* __restrict__ B,
                                                  const float* __restrict__ bias,
                                                  float* __restrict__ C,
                                                  int M, int N, int K, int act) {
  __shared__ __align__(16) _Float16 Als[64 * 32];
  __shared__ __align__(16) _Float16 Bls[64 * 32];

  const int tid  = threadIdx.x;
  const int lane = tid & 31;
  const int wave = tid >> 5;
  const int mBase = blockIdx.y * 64;
  const int nBase = blockIdx.x * 64;
  const int mi  = wave & 3;          // m-tile 0..3
  const int nj0 = (wave >> 2) * 2;   // n-tiles nj0, nj0+1

  v8f acc0 = {};
  v8f acc1 = {};

  const int lrow = tid >> 2;         // 0..63
  const int lseg = (tid & 3) * 8;    // 0,8,16,24 (halves)

  const int nc    = lane & 15;
  const int khsel = (lane >> 4) & 1;
  const int mloc  = mi * 16 + nc;

  for (int k0 = 0; k0 < K; k0 += 32) {
    // ---- stage A/B tiles into LDS (16B vector loads) ----
    {
      int gm = mBase + lrow;
      uint4 av;
      if (gm < M) {
        av = *(const uint4*)(A + (size_t)gm * K + k0 + lseg);
      } else {
        av.x = av.y = av.z = av.w = 0u;
      }
      *(uint4*)&Als[lrow * 32 + lseg] = av;
      int gn = nBase + lrow;  // always in range (N % 64 == 0)
      *(uint4*)&Bls[lrow * 32 + lseg] =
          *(const uint4*)(B + (size_t)gn * K + k0 + lseg);
      // prefetch next K-chunk (lowers to global_prefetch_b8)
      if (k0 + 32 < K) {
        if (gm < M) __builtin_prefetch(A + (size_t)gm * K + k0 + 32 + lseg, 0, 1);
        __builtin_prefetch(B + (size_t)gn * K + k0 + 32 + lseg, 0, 1);
      }
    }
    __syncthreads();

    // ---- build fragments per documented 16-bit A/B VGPR layouts ----
    Frag16 a, b0, b1;
    const unsigned* arow  = (const unsigned*)&Als[mloc * 32];
    const unsigned* brow0 = (const unsigned*)&Bls[(nj0 * 16 + nc) * 32];
    const unsigned* brow1 = (const unsigned*)&Bls[((nj0 + 1) * 16 + nc) * 32];
#pragma unroll
    for (int i = 0; i < 8; ++i) {
      int ua = (i & 3) + khsel * 4 + ((i >> 2) << 3);  // A pattern
      int ub = (i & 3) + khsel * 4 + ((i >> 2) << 3);  // B pattern (same as A)
      a.u[i]  = arow[ua];
      b0.u[i] = brow0[ub];
      b1.u[i] = brow1[ub];
    }
    acc0 = __builtin_amdgcn_wmma_f32_16x16x32_f16(false, a.v, false, b0.v,
                                                  (short)0, acc0, false, false);
    acc1 = __builtin_amdgcn_wmma_f32_16x16x32_f16(false, a.v, false, b1.v,
                                                  (short)0, acc1, false, false);
    __syncthreads();
  }

  // ---- store D (C/D layout: reg r, lanes 0-15 -> M=r, lanes 16-31 -> M=r+8)
  const int msel = (lane >> 4) * 8;
#pragma unroll
  for (int r = 0; r < 8; ++r) {
    int m = mBase + mi * 16 + msel + r;
    if (m < M) {
      int n0 = nBase + nj0 * 16 + nc;
      int n1 = n0 + 16;
      float v0 = acc0[r] + (bias ? bias[n0] : 0.0f);
      float v1 = acc1[r] + (bias ? bias[n1] : 0.0f);
      if (act) { v0 = tanhf(v0); v1 = tanhf(v1); }
      C[(size_t)m * N + n0] = v0;
      C[(size_t)m * N + n1] = v1;
    }
  }
}

// --------------------------- fused GRU step --------------------------------
// One workgroup handles 16 sequence rows. Computes hg = h@Whh^T via WMMA
// (staged in LDS), then gate math; updates h (f32) + h16 (f16) in place and
// optionally records h16 into an output sequence buffer.
// xg layout: [seq*L + t, 768]. hidden = 256. 256 threads = 8 waves.

__global__ __launch_bounds__(256) void k_gru_step(const float* __restrict__ xg,
                                                  int t, int L,
                                                  float* __restrict__ h,
                                                  _Float16* __restrict__ h16,
                                                  const _Float16* __restrict__ Whh,
                                                  const float* __restrict__ bhh,
                                                  const int* __restrict__ mask,
                                                  _Float16* __restrict__ outseq,
                                                  int outT) {
  __shared__ __align__(16) _Float16 Hls[16 * 256];  // h16 tile (A matrix)
  __shared__ float Hg[16 * 768];                    // hg accumulators

  const int tid  = threadIdx.x;
  const int lane = tid & 31;
  const int wave = tid >> 5;
  const int rowBase = blockIdx.x * 16;

  // stage h16 (16 x 256 halves) into LDS
  {
    int r = tid >> 4;
    int c = (tid & 15) * 16;
    const _Float16* src = h16 + (size_t)(rowBase + r) * 256 + c;
    *(uint4*)&Hls[r * 256 + c]     = *(const uint4*)(src);
    *(uint4*)&Hls[r * 256 + c + 8] = *(const uint4*)(src + 8);
  }
  __syncthreads();

  const int nc    = lane & 15;
  const int khsel = (lane >> 4) & 1;
  const int msel  = (lane >> 4) * 8;

  // 48 n-tiles of 16 cols; 6 per wave; K=256 in 8 chunks of 32
  for (int nt = wave * 6; nt < wave * 6 + 6; ++nt) {
    v8f acc = {};
    for (int kc = 0; kc < 256; kc += 32) {
      Frag16 a, b;
      const unsigned* arow = (const unsigned*)&Hls[nc * 256 + kc];
      const unsigned* brow =
          (const unsigned*)(Whh + (size_t)(nt * 16 + nc) * 256 + kc);
#pragma unroll
      for (int i = 0; i < 8; ++i) {
        int ua = (i & 3) + khsel * 4 + ((i >> 2) << 3);  // A pattern
        int ub = (i & 3) + khsel * 4 + ((i >> 2) << 3);  // B pattern
        a.u[i] = arow[ua];
        b.u[i] = brow[ub];
      }
      acc = __builtin_amdgcn_wmma_f32_16x16x32_f16(false, a.v, false, b.v,
                                                   (short)0, acc, false, false);
    }
#pragma unroll
    for (int r = 0; r < 8; ++r)
      Hg[(msel + r) * 768 + nt * 16 + nc] = acc[r];
  }
  __syncthreads();

  // gate math: one thread per hidden column
  const int j = tid;
  const float bh_r = bhh[j];
  const float bh_z = bhh[256 + j];
  const float bh_n = bhh[512 + j];
  for (int row = 0; row < 16; ++row) {
    int seq = rowBase + row;
    const float* xrow = xg + ((size_t)seq * L + t) * 768;
    float xr = xrow[j];
    float xz = xrow[256 + j];
    float xn = xrow[512 + j];
    float hr = Hg[row * 768 + j]       + bh_r;
    float hz = Hg[row * 768 + 256 + j] + bh_z;
    float hn = Hg[row * 768 + 512 + j] + bh_n;
    float rg = sigmoidf_(xr + hr);
    float zg = sigmoidf_(xz + hz);
    float ng = tanhf(xn + rg * hn);
    size_t hidx = (size_t)seq * 256 + j;
    float hprev = h[hidx];
    float hnew  = (1.0f - zg) * ng + zg * hprev;
    if (mask && mask[(size_t)seq * L + t] == 0) hnew = hprev;
    h[hidx]   = hnew;
    h16[hidx] = (_Float16)hnew;
    if (outseq) outseq[((size_t)seq * outT + t) * 256 + j] = (_Float16)hnew;
  }
}

// --------------------------- memory attention scan -------------------------
// One block per batch element; mem[16][256] stays resident in LDS for all
// 32 steps. Writes z2 (f32) and z2 (f16).

__global__ __launch_bounds__(256) void k_mem_scan(const float* __restrict__ z,
                                                  const float* __restrict__ mem_init,
                                                  float* __restrict__ z2,
                                                  _Float16* __restrict__ z2h) {
  __shared__ float memL[16 * 256];
  __shared__ float qL[256];
  __shared__ float part[16 * 16];
  __shared__ float wL[16];

  const int b   = blockIdx.x;
  const int tid = threadIdx.x;

  for (int i = tid; i < 16 * 256; i += 256) memL[i] = mem_init[i];
  __syncthreads();

  for (int k = 0; k < 32; ++k) {
    qL[tid] = z[((size_t)b * 32 + k) * 256 + tid];
    __syncthreads();

    {  // partial dot products: thread (m = tid/16, p = tid%16)
      int m = tid >> 4, p = tid & 15;
      float s = 0.0f;
      const float* mr = &memL[m * 256];
      for (int jj = p * 16; jj < p * 16 + 16; ++jj) s += mr[jj] * qL[jj];
      part[m * 16 + p] = s;
    }
    __syncthreads();

    if (tid == 0) {  // softmax over 16 scores (tiny)
      float sc[16];
      float mx = -1e30f;
      for (int m = 0; m < 16; ++m) {
        float s = 0.0f;
        for (int p = 0; p < 16; ++p) s += part[m * 16 + p];
        s *= 0.0625f;  // 1/sqrt(256)
        sc[m] = s;
        mx = fmaxf(mx, s);
      }
      float den = 0.0f;
      for (int m = 0; m < 16; ++m) { sc[m] = __expf(sc[m] - mx); den += sc[m]; }
      float inv = 1.0f / den;
      for (int m = 0; m < 16; ++m) wL[m] = sc[m] * inv;
    }
    __syncthreads();

    {  // readout (old mem) + memory update, one thread per column
      const int jj = tid;
      float qj = qL[jj];
      float ro = 0.0f;
      for (int m = 0; m < 16; ++m) {
        float w  = wL[m];
        float mv = memL[m * 256 + jj];
        ro += w * mv;
        memL[m * 256 + jj] = mv * (1.0f - w) + w * qj;
      }
      float outv = qj + ro;
      z2[((size_t)b * 32 + k) * 256 + jj]  = outv;
      z2h[((size_t)b * 32 + k) * 256 + jj] = (_Float16)outv;
    }
    __syncthreads();
  }
}

// --------------------------- host orchestration ----------------------------

extern "C" void kernel_launch(void* const* d_in, const int* in_sizes, int n_in,
                              void* d_out, int out_size, void* d_ws, size_t ws_size,
                              hipStream_t stream) {
  (void)in_sizes; (void)n_in; (void)out_size; (void)ws_size;

  const int* ctx_ids   = (const int*)d_in[0];
  const int* ctx_mask  = (const int*)d_in[1];
  const float* img     = (const float*)d_in[2];
  const int* tgt       = (const int*)d_in[3];
  const float* emb_txt = (const float*)d_in[4];
  const float* Wih_t   = (const float*)d_in[5];
  const float* Whh_t   = (const float*)d_in[6];
  const float* bih_t   = (const float*)d_in[7];
  const float* bhh_t   = (const float*)d_in[8];
  const float* fuse_W  = (const float*)d_in[9];
  const float* fuse_b  = (const float*)d_in[10];
  const float* mem0    = (const float*)d_in[11];
  const float* Wih_c   = (const float*)d_in[12];
  const float* Whh_c   = (const float*)d_in[13];
  const float* bih_c   = (const float*)d_in[14];
  const float* bhh_c   = (const float*)d_in[15];
  const float* tok_emb = (const float*)d_in[16];
  const float* Wih_d   = (const float*)d_in[17];
  const float* Whh_d   = (const float*)d_in[18];
  const float* bih_d   = (const float*)d_in[19];
  const float* bhh_d   = (const float*)d_in[20];
  const float* head_W  = (const float*)d_in[21];
  const float* head_b  = (const float*)d_in[22];
  float* out = (float*)d_out;

  enum { V = 32000, DM = 256, IMG = 512, H = 256, B = 16, K = 32, L = 32, T = 128 };
  const int BK   = B * K;        // 512
  const int NTXT = BK * L;       // 16384
  const int NDEC = B * (T - 1);  // 2032
  const int WSZ  = 3 * DM * DM;  // 196608 (per GRU weight matrix)

  // -------- workspace carve-out --------
  size_t off = 0;
  char* base = (char*)d_ws;
  auto alloc = [&](size_t bytes) -> void* {
    void* p = base + off;
    off += (bytes + 255) & ~(size_t)255;
    return p;
  };
  _Float16* WihT  = (_Float16*)alloc((size_t)WSZ * 2);
  _Float16* WhhT  = (_Float16*)alloc((size_t)WSZ * 2);
  _Float16* WihC  = (_Float16*)alloc((size_t)WSZ * 2);
  _Float16* WhhC  = (_Float16*)alloc((size_t)WSZ * 2);
  _Float16* WihD  = (_Float16*)alloc((size_t)WSZ * 2);
  _Float16* WhhD  = (_Float16*)alloc((size_t)WSZ * 2);
  _Float16* fuseWh = (_Float16*)alloc((size_t)H * (DM + IMG) * 2);
  _Float16* headWh = (_Float16*)alloc((size_t)V * H * 2);
  _Float16* tok_h  = (_Float16*)alloc((size_t)NTXT * DM * 2);
  float*    xg     = (float*)   alloc((size_t)NTXT * 3 * DM * 4);  // reused 3x
  float*    h_t    = (float*)   alloc((size_t)BK * DM * 4);
  _Float16* h_t_h  = (_Float16*)alloc((size_t)BK * DM * 2);
  _Float16* fin    = (_Float16*)alloc((size_t)BK * (DM + IMG) * 2);
  float*    zbuf   = (float*)   alloc((size_t)BK * H * 4);
  float*    z2     = (float*)   alloc((size_t)BK * H * 4);
  _Float16* z2h    = (_Float16*)alloc((size_t)BK * H * 2);
  float*    h_c    = (float*)   alloc((size_t)B * H * 4);
  _Float16* h_c_h  = (_Float16*)alloc((size_t)B * H * 2);
  _Float16* x_h    = (_Float16*)alloc((size_t)NDEC * H * 2);
  _Float16* outs_h = (_Float16*)alloc((size_t)NDEC * H * 2);

  auto cvt = [&](const float* s, _Float16* d, int n) {
    k_cvt_f16<<<(n + 255) / 256, 256, 0, stream>>>(s, d, n);
  };

  // -------- weight conversions --------
  cvt(Wih_t, WihT, WSZ);  cvt(Whh_t, WhhT, WSZ);
  cvt(Wih_c, WihC, WSZ);  cvt(Whh_c, WhhC, WSZ);
  cvt(Wih_d, WihD, WSZ);  cvt(Whh_d, WhhD, WSZ);
  cvt(fuse_W, fuseWh, H * (DM + IMG));
  cvt(head_W, headWh, V * H);

  // -------- text GRU --------
  {
    int n = NTXT * DM;
    k_gather_f16<<<(n + 255) / 256, 256, 0, stream>>>(ctx_ids, emb_txt, tok_h, n, DM);
  }
  k_gemm_f16<<<dim3(3 * DM / 64, (NTXT + 63) / 64), 256, 0, stream>>>(
      tok_h, WihT, bih_t, xg, NTXT, 3 * DM, DM, 0);
  k_fill0_f32<<<(BK * DM + 255) / 256, 256, 0, stream>>>(h_t, BK * DM);
  k_fill0_f16<<<(BK * DM + 255) / 256, 256, 0, stream>>>(h_t_h, BK * DM);
  for (int t = 0; t < L; ++t) {
    k_gru_step<<<BK / 16, 256, 0, stream>>>(xg, t, L, h_t, h_t_h, WhhT, bhh_t,
                                            ctx_mask, (_Float16*)nullptr, 0);
  }

  // -------- fuse + memory scan --------
  {
    int n = BK * (DM + IMG);
    k_build_fin<<<(n + 255) / 256, 256, 0, stream>>>(h_t_h, img, fin, BK, DM, IMG);
  }
  k_gemm_f16<<<dim3(H / 64, (BK + 63) / 64), 256, 0, stream>>>(
      fin, fuseWh, fuse_b, zbuf, BK, H, DM + IMG, 1 /*tanh*/);
  k_mem_scan<<<B, 256, 0, stream>>>(zbuf, mem0, z2, z2h);

  // -------- context GRU --------
  k_gemm_f16<<<dim3(3 * H / 64, (BK + 63) / 64), 256, 0, stream>>>(
      z2h, WihC, bih_c, xg, BK, 3 * H, H, 0);
  k_fill0_f32<<<(B * H + 255) / 256, 256, 0, stream>>>(h_c, B * H);
  k_fill0_f16<<<(B * H + 255) / 256, 256, 0, stream>>>(h_c_h, B * H);
  for (int t = 0; t < K; ++t) {
    k_gru_step<<<1, 256, 0, stream>>>(xg, t, K, h_c, h_c_h, WhhC, bhh_c,
                                      (const int*)nullptr, (_Float16*)nullptr, 0);
  }

  // -------- decoder GRU (h0 = h_c, updated in place) --------
  {
    int n = NDEC * H;
    k_gather_tgt_f16<<<(n + 255) / 256, 256, 0, stream>>>(tgt, tok_emb, x_h,
                                                          T - 1, T, H, n);
  }
  k_gemm_f16<<<dim3(3 * H / 64, (NDEC + 63) / 64), 256, 0, stream>>>(
      x_h, WihD, bih_d, xg, NDEC, 3 * H, H, 0);
  for (int t = 0; t < T - 1; ++t) {
    k_gru_step<<<1, 256, 0, stream>>>(xg, t, T - 1, h_c, h_c_h, WhhD, bhh_d,
                                      (const int*)nullptr, outs_h, T - 1);
  }

  // -------- head GEMM: (2032 x 256) @ (32000 x 256)^T + head_b --------
  k_gemm_f16<<<dim3(V / 64, (NDEC + 63) / 64), 256, 0, stream>>>(
      outs_h, headWh, head_b, out, NDEC, V, H, 0);
}
